// VanillaGCNFeatureEmbedding_40037685133336
// MI455X (gfx1250) — matrix operand order repaired and checked by reading, not compiled, not run on hardware
//
#include <hip/hip_runtime.h>
#include <stdint.h>

// ---------------------------------------------------------------------------
// GCN 2-layer forward on MI455X (gfx1250), bf16 WMMA path + async-to-LDS
// staging of the normalized adjacency stream.
//   An = D^-1/2 (A^T + I) D^-1/2   (materialized bf16, pre-swizzled A-fragments)
//   H1 = relu(An @ (feat @ W1) + b1)
//   out = relu(An @ (H1 @ W2) + b2)
// ---------------------------------------------------------------------------

#define NN 8192
#define FH 128

typedef __bf16 bf16_t;
typedef bf16_t v16bf __attribute__((ext_vector_type(16)));
typedef float  v8f   __attribute__((ext_vector_type(8)));

static __device__ __forceinline__ unsigned short f2bf(float f) {
    unsigned int u = __float_as_uint(f);
    u += 0x7FFFu + ((u >> 16) & 1u);   // round-to-nearest-even
    return (unsigned short)(u >> 16);
}

// --------------------------- async-to-LDS support --------------------------
#if defined(__AMDGCN__) && __has_builtin(__builtin_amdgcn_global_load_async_to_lds_b128)
#define USE_ASYNC 1
#else
#define USE_ASYNC 0
#endif

#if USE_ASYNC
// Builtin expects: (int4 AS1*, int4 AS3*, imm offset, imm cpol)
// (per hipcc diagnostic: param type is '__vector_size__(16) int __device__ *').
typedef int v4i_vs __attribute__((vector_size(16)));
typedef __attribute__((address_space(1))) v4i_vs* as1_v4i_ptr;
typedef __attribute__((address_space(3))) v4i_vs* as3_v4i_ptr;

// generic->AS1 keeps the full 64-bit address; generic->AS3 truncates to the
// 32-bit LDS offset (ISA 10.2: the low 32 address bits).
static __device__ __forceinline__ void async_cp16(const unsigned short* g,
                                                  unsigned short* l) {
    __builtin_amdgcn_global_load_async_to_lds_b128(
        (as1_v4i_ptr)(uintptr_t)g,
        (as3_v4i_ptr)(unsigned int)(uintptr_t)l,
        0, 0);
}
#if __has_builtin(__builtin_amdgcn_s_wait_asynccnt)
#define WAIT_ASYNC(n) __builtin_amdgcn_s_wait_asynccnt(n)
#else
#define WAIT_ASYNC(n) asm volatile("s_wait_asynccnt %0" ::"n"(n) : "memory")
#endif
#endif

// ---------------------------------------------------------------------------
// Kernel 1: partial column sums of A.  grid (32 colblocks, 32 rowslices) x 256
// ---------------------------------------------------------------------------
__global__ __launch_bounds__(256) void colsum_kernel(
    const float* __restrict__ A, float* __restrict__ partial)
{
    int i  = blockIdx.x * 256 + threadIdx.x;
    int j0 = blockIdx.y * 256;
    float s = 0.0f;
    #pragma unroll 4
    for (int jj = 0; jj < 256; ++jj) {
        s += A[(size_t)(j0 + jj) * NN + i];
    }
    partial[(size_t)blockIdx.y * NN + i] = s;
}

// ---------------------------------------------------------------------------
// Kernel 2: dis[i] = rsqrt(1 + colsum_i)   (deg >= 1 always, matches reference)
// ---------------------------------------------------------------------------
__global__ __launch_bounds__(256) void mkdis_kernel(
    const float* __restrict__ partial, float* __restrict__ dis)
{
    int i = blockIdx.x * 256 + threadIdx.x;
    float s = 1.0f;
    #pragma unroll
    for (int r = 0; r < 32; ++r) s += partial[(size_t)r * NN + i];
    dis[i] = rsqrtf(s);
}

// ---------------------------------------------------------------------------
// Kernel 3: materialize An in bf16, pre-swizzled into WMMA A-operand layout.
// A-fragment tile = 16 rows (M) x 32 cols (K), 512 bf16 contiguous:
//   offset = lane*16 + el,  lane = (M&15) + 16*hi,
//   K(el,hi) = el + 16*hi
// Tiles k-major per 16-row strip: tile(TI,TK) at (TI*256+TK)*512.
// ---------------------------------------------------------------------------
__global__ __launch_bounds__(256) void mat_an_kernel(
    const float* __restrict__ A, const float* __restrict__ dis,
    unsigned short* __restrict__ An)
{
    __shared__ float tile[64 * 65];
    int i0 = blockIdx.x * 64;   // An row block
    int j0 = blockIdx.y * 64;   // An col block
    int tid = threadIdx.x;
    int il = tid & 63;
    int jq = tid >> 6;

    float di = dis[i0 + il];
    #pragma unroll 4
    for (int rr = 0; rr < 16; ++rr) {
        int jl = rr * 4 + jq;
        int i = i0 + il;
        int j = j0 + jl;
        float a = A[(size_t)j * NN + i];
        if (i == j) a += 1.0f;
        tile[il * 65 + jl] = di * dis[j] * a;
    }
    __syncthreads();

    int TI0 = blockIdx.x * 4;
    int TK0 = blockIdx.y * 2;
    #pragma unroll
    for (int s = 0; s < 16; ++s) {
        int t  = s >> 1;
        int ti = t & 3;
        int tk = t >> 2;
        int e  = ((s & 1) << 8) + tid;
        int lane = e >> 4, el = e & 15;
        int m  = lane & 15, hi = lane >> 4;
        int k  = el + 16 * hi;
        float v = tile[(ti * 16 + m) * 65 + (tk * 32 + k)];
        An[((size_t)(TI0 + ti) * (NN / 32) + (TK0 + tk)) * 512 + e] = f2bf(v);
    }
}

// ---------------------------------------------------------------------------
// Kernel 4: Xb = (X @ W) as bf16, pre-swizzled into WMMA B-operand layout.
// B tile = 32 rows (K) x 16 cols (N): lane = (N&15) + 16*(K>=16), el = K&15.
// ---------------------------------------------------------------------------
__global__ __launch_bounds__(256) void xform_kernel(
    const float* __restrict__ X, const float* __restrict__ W,
    unsigned short* __restrict__ Xb)
{
    __shared__ float Wl[FH * FH];
    int tid = threadIdx.x;
    for (int idx = tid; idx < FH * FH; idx += 256) Wl[idx] = W[idx];
    __syncthreads();

    int r  = blockIdx.x * 32 + (tid >> 3);
    int c0 = (tid & 7) << 4;

    const float* xrow = X + (size_t)r * FH;
    float acc[16];
    #pragma unroll
    for (int j = 0; j < 16; ++j) acc[j] = 0.0f;

    for (int c = 0; c < FH; ++c) {
        float xv = xrow[c];
        const float* wr = Wl + c * FH + c0;
        #pragma unroll
        for (int j = 0; j < 16; ++j) acc[j] += xv * wr[j];
    }

    int kt = r >> 5, kl = r & 31;
    int hi = kl >> 4, e = kl & 15;
    int nt = c0 >> 4;
    unsigned short* bp = Xb + ((size_t)kt * 8 + nt) * 512 + e;
    #pragma unroll
    for (int j = 0; j < 16; ++j) bp[(j + 16 * hi) * 16] = f2bf(acc[j]);
}

// ---------------------------------------------------------------------------
// Kernel 5: out = relu(An @ X + bias), bf16 WMMA, fp32 accumulate.
// grid (128, 2) x 256 threads (8 waves).  Wave w: row-tile rt = w&3,
// col-quarter cq = w>>2 (two 16-col tiles).
// An A-fragments are staged once per block through LDS with async copies
// (double buffered, 2 k-tiles per stage) when the toolchain exposes them.
// ---------------------------------------------------------------------------
__global__ __launch_bounds__(256) void gcn_spmm_kernel(
    const unsigned short* __restrict__ An,
    const unsigned short* __restrict__ Xb,
    const float* __restrict__ bias,
    float* __restrict__ out)
{
    int tid  = threadIdx.x;
    int w    = tid >> 5;
    int lane = tid & 31;
    int rt = w & 3, cq = w >> 2;

    int TI0 = blockIdx.x * 4;
    int TI  = TI0 + rt;
    int nt0 = blockIdx.y * 4 + cq * 2;

    const v16bf* Bv = (const v16bf*)Xb;

    v8f acc0 = {0.f, 0.f, 0.f, 0.f, 0.f, 0.f, 0.f, 0.f};
    v8f acc1 = {0.f, 0.f, 0.f, 0.f, 0.f, 0.f, 0.f, 0.f};

#if USE_ASYNC
    // 2 buffers x (2 k-steps x 4 row-tiles x 512 bf16) = 16 KB
    __shared__ alignas(64) unsigned short Asm[2][2 * 4 * 512];

    int srt  = tid >> 6;           // staging row-tile 0..3
    int soff = (tid & 63) * 8;     // staging element offset (16 B per thread)
    const unsigned short* Ag =
        An + (size_t)(TI0 + srt) * (NN / 32) * 512 + soff;

    // prologue: stage k-tiles 0,1 into buffer 0
    async_cp16(Ag,       &Asm[0][0 * 2048 + srt * 512 + soff]);
    async_cp16(Ag + 512, &Asm[0][1 * 2048 + srt * 512 + soff]);

    for (int kt2 = 0; kt2 < NN / 64; ++kt2) {
        int buf = kt2 & 1;
        if (kt2 + 1 < NN / 64) {
            const unsigned short* An_next = Ag + (size_t)(2 * kt2 + 2) * 512;
            async_cp16(An_next,       &Asm[buf ^ 1][0 * 2048 + srt * 512 + soff]);
            async_cp16(An_next + 512, &Asm[buf ^ 1][1 * 2048 + srt * 512 + soff]);
            WAIT_ASYNC(2);   // current buffer's pair complete (in-order)
        } else {
            WAIT_ASYNC(0);
        }
        __syncthreads();
        #pragma unroll
        for (int s = 0; s < 2; ++s) {
            int kt = kt2 * 2 + s;
            v16bf a  = *(const v16bf*)&Asm[buf][s * 2048 + rt * 512 + lane * 16];
            v16bf b0 = Bv[((size_t)kt * 8 + nt0) * 32 + lane];
            v16bf b1 = Bv[((size_t)kt * 8 + nt0 + 1) * 32 + lane];
            acc0 = __builtin_amdgcn_wmma_f32_16x16x32_bf16(
                false, a, false, b0, (short)0, acc0, false, false);
            acc1 = __builtin_amdgcn_wmma_f32_16x16x32_bf16(
                false, a, false, b1, (short)0, acc1, false, false);
        }
        __syncthreads();   // protect buffer before next stage overwrites it
    }
#else
    const v16bf* Av = (const v16bf*)An;
    size_t abase = (size_t)TI * (NN / 32) * 32 + lane;
    #pragma unroll 2
    for (int kt = 0; kt < NN / 32; ++kt) {
        v16bf a  = Av[abase + (size_t)kt * 32];
        v16bf b0 = Bv[((size_t)kt * 8 + nt0) * 32 + lane];
        v16bf b1 = Bv[((size_t)kt * 8 + nt0 + 1) * 32 + lane];
        acc0 = __builtin_amdgcn_wmma_f32_16x16x32_bf16(
            false, a, false, b0, (short)0, acc0, false, false);
        acc1 = __builtin_amdgcn_wmma_f32_16x16x32_bf16(
            false, a, false, b1, (short)0, acc1, false, false);
    }
#endif

    // C/D layout: VGPR v -> row M = v + 8*(lane>=16); col N = lane&15
    int row0 = TI * 16 + 8 * (lane >> 4);
    int coln = lane & 15;

    {
        int col = nt0 * 16 + coln;
        float bc = bias[col];
        #pragma unroll
        for (int v = 0; v < 8; ++v) {
            float x = acc0[v] + bc;
            out[(size_t)(row0 + v) * FH + col] = fmaxf(x, 0.0f);
        }
    }
    {
        int col = (nt0 + 1) * 16 + coln;
        float bc = bias[col];
        #pragma unroll
        for (int v = 0; v < 8; ++v) {
            float x = acc1[v] + bc;
            out[(size_t)(row0 + v) * FH + col] = fmaxf(x, 0.0f);
        }
    }
}

// ---------------------------------------------------------------------------
// Host launcher.
// ws layout:  [0,32K)        dis (8192 f32)
//             [32K, 32K+4M)  H1  (8192x128 f32)   (reused as colsum partials)
//             [+2M)          Xb  (8192x128 bf16, B-swizzled)
//             [+128M)        An  (8192x8192 bf16, A-swizzled)
// ---------------------------------------------------------------------------
extern "C" void kernel_launch(void* const* d_in, const int* in_sizes, int n_in,
                              void* d_out, int out_size, void* d_ws, size_t ws_size,
                              hipStream_t stream) {
    (void)in_sizes; (void)n_in; (void)out_size; (void)ws_size;

    const float* feat = (const float*)d_in[0];
    const float* A    = (const float*)d_in[1];
    const float* W1   = (const float*)d_in[2];
    const float* b1   = (const float*)d_in[3];
    const float* W2   = (const float*)d_in[4];
    const float* b2   = (const float*)d_in[5];
    float* out = (float*)d_out;

    char* ws = (char*)d_ws;
    float*          dis = (float*)ws;
    float*          H1  = (float*)(ws + 32768);
    unsigned short* Xb  = (unsigned short*)(ws + 32768 + 4194304);
    unsigned short* An  = (unsigned short*)(ws + 32768 + 4194304 + 2097152);
    float*          partial = H1;   // 1 MB, dead before H1 is written

    // 1) degrees -> dis
    colsum_kernel<<<dim3(32, 32), 256, 0, stream>>>(A, partial);
    mkdis_kernel<<<32, 256, 0, stream>>>(partial, dis);

    // 2) An (transposed, normalized, bf16, A-fragment swizzled)
    mat_an_kernel<<<dim3(128, 128), 256, 0, stream>>>(A, dis, An);

    // 3) layer 1
    xform_kernel<<<NN / 32, 256, 0, stream>>>(feat, W1, Xb);
    gcn_spmm_kernel<<<dim3(128, 2), 256, 0, stream>>>(An, Xb, b1, H1);

    // 4) layer 2
    xform_kernel<<<NN / 32, 256, 0, stream>>>(H1, W2, Xb);
    gcn_spmm_kernel<<<dim3(128, 2), 256, 0, stream>>>(An, Xb, b2, out);
}
